// DeepSeek_MoE_79078937854406
// MI455X (gfx1250) — matrix operand order repaired and checked by reading, not compiled, not run on hardware
//
#include <hip/hip_runtime.h>
#include <cstdint>
#include <cstddef>

typedef __bf16 bf16;
typedef __attribute__((ext_vector_type(4)))  __bf16 v4bf;
typedef __attribute__((ext_vector_type(8)))  __bf16 v8bf;
typedef __attribute__((ext_vector_type(16))) __bf16 v16bf;
typedef __attribute__((ext_vector_type(8)))  float  v8f;

#define T_TOK 8192
#define HDIM  1024
#define LDIM  512
#define NEXP  16

#define BM 128
#define BN 64
#define BK 32
#define LDA 40   // halves per A-tile row in LDS (padded)
#define LDB 40   // halves per B-tile column in LDS (padded)

__device__ __forceinline__ float silu_f(float x) { return x / (1.f + __expf(-x)); }

__device__ __forceinline__ v8f wmma_bf16(v16bf a, v16bf b, v8f c) {
  return __builtin_amdgcn_wmma_f32_16x16x32_bf16(false, a, false, b, (short)0, c, false, false);
}

// Generic->LDS byte offset (shared aperture keeps LDS offset in low 32 bits)
__device__ __forceinline__ unsigned lds_off_u32(const void* p) {
  return (unsigned)(unsigned long long)p;
}
// Async LDS-DMA: copies 16B per active lane, tracked by ASYNCcnt
__device__ __forceinline__ void async_b128(unsigned lds, const bf16* g) {
  asm volatile("global_load_async_to_lds_b128 %0, %1, off"
               :: "v"(lds), "v"(g) : "memory");
}
__device__ __forceinline__ void wait_async_le3() { asm volatile("s_wait_asynccnt 0x3" ::: "memory"); }
__device__ __forceinline__ void wait_async_0()  { asm volatile("s_wait_asynccnt 0x0" ::: "memory"); }

// A fragment: lane<16 -> row=lane, K {0..7,16..23}; lane>=16 -> row=lane-16, K {8..15,24..31}
__device__ __forceinline__ v16bf frag_a(const bf16* lA, int row, int hsel) {
  const bf16* p = lA + row * LDA + hsel * 8;
  v8bf lo = *(const v8bf*)p;
  v8bf hi = *(const v8bf*)(p + 16);
  return __builtin_shufflevector(lo, hi, 0,1,2,3,4,5,6,7,8,9,10,11,12,13,14,15);
}
// B fragment: lane<16 -> col=lane, K {0..7,16..23}; lane>=16 -> col=lane-16, K {8..15,24..31} (lB stored [n][k])
__device__ __forceinline__ v16bf frag_b(const bf16* lB, int col, int hsel) {
  const bf16* p = lB + col * LDB + hsel * 8;
  v8bf lo = *(const v8bf*)p;
  v8bf hi = *(const v8bf*)(p + 16);
  return __builtin_shufflevector(lo, hi, 0,1,2,3,4,5,6,7,8,9,10,11,12,13,14,15);
}

__device__ __forceinline__ void mma_step(const bf16* lA, const bf16* lB,
                                         int wm, int wn, int l15, int hsel,
                                         v8f acc[2][2]) {
  v16bf a0 = frag_a(lA, wm * 32 + l15, hsel);
  v16bf a1 = frag_a(lA, wm * 32 + 16 + l15, hsel);
  v16bf b0 = frag_b(lB, wn * 32 + l15, hsel);
  v16bf b1 = frag_b(lB, wn * 32 + 16 + l15, hsel);
  acc[0][0] = wmma_bf16(a0, b0, acc[0][0]);
  acc[0][1] = wmma_bf16(a0, b1, acc[0][1]);
  acc[1][0] = wmma_bf16(a1, b0, acc[1][0]);
  acc[1][1] = wmma_bf16(a1, b1, acc[1][1]);
}

// ---------------- conversions ----------------
__global__ __launch_bounds__(256) void convert_x(const float* __restrict__ x,
                                                 bf16* __restrict__ y) {
  size_t i = ((size_t)blockIdx.x * 256 + threadIdx.x) * 4;
  float4 v = *(const float4*)(x + i);
  v4bf o = { (bf16)v.x, (bf16)v.y, (bf16)v.z, (bf16)v.w };
  *(v4bf*)(y + i) = o;
}

// W [K][N] f32 (z-indexed expert)  ->  Wt [N][K] bf16
__global__ __launch_bounds__(256) void convert_w_t(const float* __restrict__ W,
                                                   bf16* __restrict__ Wt,
                                                   int K, int N) {
  const float* Wp = W + (size_t)blockIdx.z * K * N;
  bf16* Wtp = Wt + (size_t)blockIdx.z * K * N;
  int n  = blockIdx.x * 64 + (threadIdx.x & 63);
  int k0 = blockIdx.y * 32 + (threadIdx.x >> 6) * 8;
  v8bf o;
#pragma unroll
  for (int kk = 0; kk < 8; ++kk) o[kk] = (bf16)Wp[(size_t)(k0 + kk) * N + n];
  *(v8bf*)(Wtp + (size_t)n * K + k0) = o;
}

// ---------------- router ----------------
__global__ __launch_bounds__(256) void router_kernel(
    const float* __restrict__ flat, const float* __restrict__ cent,
    const float* __restrict__ ebias, int* __restrict__ topk_idx,
    float* __restrict__ topk_gate, int* __restrict__ counts,
    float* __restrict__ sumP) {
  __shared__ float sP[NEXP];
  __shared__ int   sC[NEXP];
  int tid = threadIdx.x;
  if (tid < NEXP) { sP[tid] = 0.f; sC[tid] = 0; }
  __syncthreads();
  int wid = tid >> 5, lane = tid & 31;
  int t = blockIdx.x * 8 + wid;

  float acc[NEXP];
#pragma unroll
  for (int e = 0; e < NEXP; ++e) acc[e] = 0.f;
  for (int h = lane; h < HDIM; h += 32) {
    float xv = flat[(size_t)t * HDIM + h];
#pragma unroll
    for (int e = 0; e < NEXP; ++e) acc[e] += xv * cent[e * HDIM + h];
  }
#pragma unroll
  for (int e = 0; e < NEXP; ++e) {
    for (int off = 16; off >= 1; off >>= 1) acc[e] += __shfl_xor(acc[e], off, 32);
  }
  if (lane == 0) {
    float aff[NEXP]; float s = 0.f;
#pragma unroll
    for (int e = 0; e < NEXP; ++e) { aff[e] = 1.f / (1.f + __expf(-acc[e])); s += aff[e]; }
    float best = -1e30f; int i1 = 0;
#pragma unroll
    for (int e = 0; e < NEXP; ++e) { float bv = aff[e] + ebias[e]; if (bv > best) { best = bv; i1 = e; } }
    float best2 = -1e30f; int i2 = 0;
#pragma unroll
    for (int e = 0; e < NEXP; ++e) {
      if (e != i1) { float bv = aff[e] + ebias[e]; if (bv > best2) { best2 = bv; i2 = e; } }
    }
    float a1v = 0.f, a2v = 0.f;
#pragma unroll
    for (int e = 0; e < NEXP; ++e) { a1v = (e == i1) ? aff[e] : a1v; a2v = (e == i2) ? aff[e] : a2v; }
    float denom = a1v + a2v + 1e-8f;
    topk_idx[2 * t]      = i1;  topk_idx[2 * t + 1]  = i2;
    topk_gate[2 * t]     = a1v / denom;
    topk_gate[2 * t + 1] = a2v / denom;
    atomicAdd(&sC[i1], 1); atomicAdd(&sC[i2], 1);
    float invs = 1.f / (s + 1e-8f);
#pragma unroll
    for (int e = 0; e < NEXP; ++e) atomicAdd(&sP[e], aff[e] * invs);
  }
  __syncthreads();
  if (tid < NEXP) { atomicAdd(&counts[tid], sC[tid]); atomicAdd(&sumP[tid], sP[tid]); }
}

// ---------------- offsets + aux loss ----------------
__global__ void offsets_aux(const int* __restrict__ counts, const float* __restrict__ sumP,
                            int* __restrict__ offsets, float* __restrict__ auxout) {
  if (threadIdx.x == 0 && blockIdx.x == 0) {
    int off = 0; float aux = 0.f;
    for (int e = 0; e < NEXP; ++e) {
      offsets[e] = off; off += counts[e];
      aux += ((float)counts[e] / 1024.f) * (sumP[e] / (float)T_TOK);
    }
    auxout[0] = 0.001f * aux;
  }
}

// ---------------- scatter tokens into per-expert row lists ----------------
__global__ __launch_bounds__(256) void scatter_kernel(
    const int* __restrict__ topk_idx, const float* __restrict__ topk_gate,
    const int* __restrict__ offsets, int* __restrict__ counts2,
    int* __restrict__ erow, float* __restrict__ rgate) {
  int t = blockIdx.x * 256 + threadIdx.x;
  if (t >= T_TOK) return;
#pragma unroll
  for (int k = 0; k < 2; ++k) {
    int e = topk_idx[2 * t + k];
    int slot = atomicAdd(&counts2[e], 1);
    int row = offsets[e] + slot;
    erow[row] = t;
    rgate[row] = topk_gate[2 * t + k];
  }
}

// ============ WMMA GEMM core (double-buffered async LDS-DMA pipeline) ============
// Per tile per wave: 3 async b128 instructions (2 for A chunk of this thread, 1 for B).
// wait asynccnt<=3 + barrier guarantees tile i fully resident (in-order completion).

#define GEMM_PREAMBLE()                                                         \
  const int tid = threadIdx.x;                                                  \
  const int wid = tid >> 5, lane = tid & 31;                                    \
  const int wm = wid >> 1, wn = wid & 1;                                        \
  const int l15 = lane & 15, hsel = lane >> 4;                                  \
  const int ar = tid >> 1, apart = tid & 1;                                     \
  const int bn = tid >> 2, bpart = tid & 3;                                     \
  unsigned aoff0 = lds_off_u32(&lA[0][ar * LDA + apart * 16]);                  \
  unsigned aoff1 = lds_off_u32(&lA[1][ar * LDA + apart * 16]);                  \
  unsigned boff0 = lds_off_u32(&lB[0][bn * LDB + bpart * 8]);                   \
  unsigned boff1 = lds_off_u32(&lB[1][bn * LDB + bpart * 8]);                   \
  v8f zz = {0.f,0.f,0.f,0.f,0.f,0.f,0.f,0.f};                                   \
  v8f acc[2][2] = {{zz, zz}, {zz, zz}};

#define GEMM_PIPELINE(NK)                                                       \
  async_b128(aoff0, abase);      async_b128(aoff0 + 16, abase + 8);             \
  async_b128(boff0, bbase);                                                     \
  async_b128(aoff1, abase + BK); async_b128(aoff1 + 16, abase + BK + 8);        \
  async_b128(boff1, bbase + BK);                                                \
  for (int i = 0; i < (NK); ++i) {                                              \
    if (i < (NK) - 1) wait_async_le3(); else wait_async_0();                    \
    __syncthreads();                                                            \
    mma_step(&lA[i & 1][0], &lB[i & 1][0], wm, wn, l15, hsel, acc);             \
    __syncthreads();                                                            \
    if (i + 2 < (NK)) {                                                         \
      int kt = (i + 2) * BK;                                                    \
      unsigned ao = (i & 1) ? aoff1 : aoff0;                                    \
      unsigned bo = (i & 1) ? boff1 : boff0;                                    \
      async_b128(ao, abase + kt); async_b128(ao + 16, abase + kt + 8);          \
      async_b128(bo, bbase + kt);                                               \
    }                                                                           \
  }

// ---------------- shared expert GEMM 1: hs = silu(X @ sw1 + sb1) ----------------
__global__ __launch_bounds__(256) void gemm_shared1(
    const bf16* __restrict__ X, const bf16* __restrict__ Wt,
    const float* __restrict__ b, bf16* __restrict__ hs) {
  __shared__ bf16 lA[2][BM * LDA];
  __shared__ bf16 lB[2][BN * LDB];
  const int rowBase = blockIdx.x * BM;
  const int colBase = blockIdx.y * BN;
  GEMM_PREAMBLE();
  const bf16* abase = X + (size_t)(rowBase + ar) * HDIM + apart * 16;
  const bf16* bbase = Wt + (size_t)(colBase + bn) * HDIM + bpart * 8;
  GEMM_PIPELINE(HDIM / BK);
#pragma unroll
  for (int fm = 0; fm < 2; ++fm)
#pragma unroll
    for (int fn = 0; fn < 2; ++fn)
#pragma unroll
      for (int iv = 0; iv < 8; ++iv) {
        int gr = rowBase + wm * 32 + fm * 16 + hsel * 8 + iv;
        int gc = colBase + wn * 32 + fn * 16 + l15;
        float x = acc[fm][fn][iv] + b[gc];
        hs[(size_t)gr * LDIM + gc] = (bf16)silu_f(x);
      }
}

// ---------------- shared expert GEMM 2: out = hidden + hs @ sw2 + sb2 ----------------
__global__ __launch_bounds__(256) void gemm_shared2(
    const bf16* __restrict__ A, const bf16* __restrict__ Wt,
    const float* __restrict__ b, const float* __restrict__ hid,
    float* __restrict__ out) {
  __shared__ bf16 lA[2][BM * LDA];
  __shared__ bf16 lB[2][BN * LDB];
  const int rowBase = blockIdx.x * BM;
  const int colBase = blockIdx.y * BN;
  GEMM_PREAMBLE();
  const bf16* abase = A + (size_t)(rowBase + ar) * LDIM + apart * 16;
  const bf16* bbase = Wt + (size_t)(colBase + bn) * LDIM + bpart * 8;
  GEMM_PIPELINE(LDIM / BK);
#pragma unroll
  for (int fm = 0; fm < 2; ++fm)
#pragma unroll
    for (int fn = 0; fn < 2; ++fn)
#pragma unroll
      for (int iv = 0; iv < 8; ++iv) {
        int gr = rowBase + wm * 32 + fm * 16 + hsel * 8 + iv;
        int gc = colBase + wn * 32 + fn * 16 + l15;
        out[(size_t)gr * HDIM + gc] =
            hid[(size_t)gr * HDIM + gc] + acc[fm][fn][iv] + b[gc];
      }
}

// ---------------- routed GEMM 1: he[row] = silu(X[token] @ rw1[e] + rb1[e]) ----------------
__global__ __launch_bounds__(256) void gemm_routed1(
    const bf16* __restrict__ X, const bf16* __restrict__ rwt1,
    const float* __restrict__ rb1, const int* __restrict__ erow,
    const int* __restrict__ counts, const int* __restrict__ offsets,
    bf16* __restrict__ he) {
  const int e = blockIdx.x >> 6;
  const int mtile = blockIdx.x & 63;
  const int cnt = counts[e];
  if (mtile * BM >= cnt) return;
  const int base = offsets[e];
  const bf16* Wt = rwt1 + (size_t)e * HDIM * LDIM;
  const float* bb = rb1 + (size_t)e * LDIM;
  __shared__ bf16 lA[2][BM * LDA];
  __shared__ bf16 lB[2][BN * LDB];
  const int colBase = blockIdx.y * BN;
  GEMM_PREAMBLE();
  // Clamp OOR rows to a valid row: uniform async issue count; garbage rows masked in epilogue
  int g = mtile * BM + ar;
  int gcl = (g < cnt) ? g : (cnt - 1);
  int tok = erow[base + gcl];
  const bf16* abase = X + (size_t)tok * HDIM + apart * 16;
  const bf16* bbase = Wt + (size_t)(colBase + bn) * HDIM + bpart * 8;
  GEMM_PIPELINE(HDIM / BK);
#pragma unroll
  for (int fm = 0; fm < 2; ++fm)
#pragma unroll
    for (int fn = 0; fn < 2; ++fn)
#pragma unroll
      for (int iv = 0; iv < 8; ++iv) {
        int gl = mtile * BM + wm * 32 + fm * 16 + hsel * 8 + iv;
        if (gl < cnt) {
          int gc = colBase + wn * 32 + fn * 16 + l15;
          float x = acc[fm][fn][iv] + bb[gc];
          he[(size_t)(base + gl) * LDIM + gc] = (bf16)silu_f(x);
        }
      }
}

// ---------------- routed GEMM 2: out[token] += gate * (he[row] @ rw2[e] + rb2[e]) ----------------
__global__ __launch_bounds__(256) void gemm_routed2(
    const bf16* __restrict__ he, const bf16* __restrict__ rwt2,
    const float* __restrict__ rb2, const int* __restrict__ erow,
    const float* __restrict__ rgate, const int* __restrict__ counts,
    const int* __restrict__ offsets, float* __restrict__ out) {
  const int e = blockIdx.x >> 6;
  const int mtile = blockIdx.x & 63;
  const int cnt = counts[e];
  if (mtile * BM >= cnt) return;
  const int base = offsets[e];
  const bf16* Wt = rwt2 + (size_t)e * LDIM * HDIM;
  const float* bb = rb2 + (size_t)e * HDIM;
  __shared__ bf16 lA[2][BM * LDA];
  __shared__ bf16 lB[2][BN * LDB];
  const int colBase = blockIdx.y * BN;
  GEMM_PREAMBLE();
  int g = mtile * BM + ar;
  int gcl = (g < cnt) ? g : (cnt - 1);
  const bf16* abase = he + (size_t)(base + gcl) * LDIM + apart * 16;
  const bf16* bbase = Wt + (size_t)(colBase + bn) * LDIM + bpart * 8;
  GEMM_PIPELINE(LDIM / BK);
#pragma unroll
  for (int fm = 0; fm < 2; ++fm)
#pragma unroll
    for (int fn = 0; fn < 2; ++fn)
#pragma unroll
      for (int iv = 0; iv < 8; ++iv) {
        int gl = mtile * BM + wm * 32 + fm * 16 + hsel * 8 + iv;
        if (gl < cnt) {
          int gc = colBase + wn * 32 + fn * 16 + l15;
          int tok = erow[base + gl];
          float gw = rgate[base + gl];
          atomicAdd(&out[(size_t)tok * HDIM + gc], gw * (acc[fm][fn][iv] + bb[gc]));
        }
      }
}

extern "C" void kernel_launch(void* const* d_in, const int* in_sizes, int n_in,
                              void* d_out, int out_size, void* d_ws, size_t ws_size,
                              hipStream_t stream) {
  const float* hidden = (const float*)d_in[0];
  const float* cent   = (const float*)d_in[1];
  const float* ebias  = (const float*)d_in[2];
  const float* sw1    = (const float*)d_in[3];
  const float* sb1    = (const float*)d_in[4];
  const float* sw2    = (const float*)d_in[5];
  const float* sb2    = (const float*)d_in[6];
  const float* rw1    = (const float*)d_in[7];
  const float* rb1    = (const float*)d_in[8];
  const float* rw2    = (const float*)d_in[9];
  const float* rb2    = (const float*)d_in[10];
  float* out = (float*)d_out;

  char* w = (char*)d_ws;
  int*   counts   = (int*)(w + 0);
  int*   counts2  = (int*)(w + 64);
  float* sumP     = (float*)(w + 128);
  int*   offsets  = (int*)(w + 192);
  size_t off = 256;
  int*   topk_idx  = (int*)(w + off);   off += (size_t)T_TOK * 2 * sizeof(int);
  float* topk_gate = (float*)(w + off); off += (size_t)T_TOK * 2 * sizeof(float);
  int*   erow      = (int*)(w + off);   off += (size_t)T_TOK * 2 * sizeof(int);
  float* rgate     = (float*)(w + off); off += (size_t)T_TOK * 2 * sizeof(float);
  bf16*  Xbf       = (bf16*)(w + off);  off += (size_t)T_TOK * HDIM * sizeof(bf16);
  bf16*  hs        = (bf16*)(w + off);  off += (size_t)T_TOK * LDIM * sizeof(bf16);
  bf16*  he        = (bf16*)(w + off);  off += (size_t)T_TOK * 2 * LDIM * sizeof(bf16);
  bf16*  swt1      = (bf16*)(w + off);  off += (size_t)HDIM * LDIM * sizeof(bf16);
  bf16*  swt2      = (bf16*)(w + off);  off += (size_t)LDIM * HDIM * sizeof(bf16);
  bf16*  rwt1      = (bf16*)(w + off);  off += (size_t)NEXP * HDIM * LDIM * sizeof(bf16);
  bf16*  rwt2      = (bf16*)(w + off);  off += (size_t)NEXP * LDIM * HDIM * sizeof(bf16);

  hipMemsetAsync(w, 0, 256, stream);
  convert_x<<<(T_TOK * HDIM) / (256 * 4), 256, 0, stream>>>(hidden, Xbf);
  // W [K][N] -> Wt [N][K] bf16
  convert_w_t<<<dim3(LDIM / 64, HDIM / 32, 1), 256, 0, stream>>>(sw1, swt1, HDIM, LDIM);
  convert_w_t<<<dim3(HDIM / 64, LDIM / 32, 1), 256, 0, stream>>>(sw2, swt2, LDIM, HDIM);
  convert_w_t<<<dim3(LDIM / 64, HDIM / 32, NEXP), 256, 0, stream>>>(rw1, rwt1, HDIM, LDIM);
  convert_w_t<<<dim3(HDIM / 64, LDIM / 32, NEXP), 256, 0, stream>>>(rw2, rwt2, LDIM, HDIM);

  router_kernel<<<T_TOK / 8, 256, 0, stream>>>(hidden, cent, ebias, topk_idx,
                                               topk_gate, counts, sumP);
  offsets_aux<<<1, 32, 0, stream>>>(counts, sumP, offsets,
                                    out + (size_t)T_TOK * HDIM);
  scatter_kernel<<<T_TOK / 256, 256, 0, stream>>>(topk_idx, topk_gate, offsets,
                                                  counts2, erow, rgate);
  gemm_shared1<<<dim3(T_TOK / BM, LDIM / BN), 256, 0, stream>>>(Xbf, swt1, sb1, hs);
  gemm_shared2<<<dim3(T_TOK / BM, HDIM / BN), 256, 0, stream>>>(hs, swt2, sb2,
                                                                hidden, out);
  gemm_routed1<<<dim3(NEXP * 64, LDIM / BN), 256, 0, stream>>>(
      Xbf, rwt1, rb1, erow, counts, offsets, he);
  gemm_routed2<<<dim3(NEXP * 64, HDIM / BN), 256, 0, stream>>>(
      he, rwt2, rb2, erow, rgate, counts, offsets, out);
}
